// BiFormerAttention_49039936586401
// MI455X (gfx1250) — matrix operand order repaired and checked by reading, not compiled, not run on hardware
//
#include <hip/hip_runtime.h>

typedef unsigned int  u32;
typedef unsigned short u16;
typedef unsigned long long u64;
typedef __attribute__((ext_vector_type(16))) __bf16 v16bf;
typedef __attribute__((ext_vector_type(2)))  __bf16 v2bf;
typedef __attribute__((ext_vector_type(8)))  float  v8f;
typedef __attribute__((ext_vector_type(8)))  u32    v8u;

#define NB 4
#define NH 12
#define NN 2048
#define CC 768
#define HD 64
#define KEEP 1024
#define BHN (NB*NH*NN)          // 98304
#define QKV_ELEMS ((size_t)NB*NH*NN*HD)  // 6291456

// ---- helpers -------------------------------------------------------------

// f32 -> bf16, round-half-up in bit domain (bias <= 2^-17 relative vs RNE).
__device__ inline u16 f2bf(float x) {
    return (u16)((__builtin_bit_cast(u32, x) + 0x8000u) >> 16);
}
// pack two f32 -> bf16x2 in one dword.
__device__ inline u32 pack2(float a, float b) {
#if __has_builtin(__builtin_amdgcn_cvt_pk_bf16_f32)
    v2bf r = __builtin_amdgcn_cvt_pk_bf16_f32(a, b);
    return __builtin_bit_cast(u32, r);
#else
    u32 ua = __builtin_bit_cast(u32, a) + 0x8000u;
    u32 ub = __builtin_bit_cast(u32, b) + 0x8000u;
    // v_perm_b32: S0=ub (bytes 4-7), S1=ua (bytes 0-3); take hi16 of each.
    return __builtin_amdgcn_perm(ub, ua, 0x07060302u);
#endif
}

// A-fragment K index (16-bit A 16x32): lanes 0-15 K=0..7,16..23 ; lanes 16-31 K=8..15,24..31
__device__ inline int kidxA(int v, int khalf) {
    return (v < 4) ? (khalf + 2 * v) : (16 + khalf + 2 * (v - 4));
}
// Contiguous K index (a code choice, not the hardware's table order): lanes 0-15 hold K=0..15, lanes 16-31 hold K=16..31, packed pairs per VGPR
__device__ inline int kidxB(int v, int hi) { return hi * 16 + 2 * v; }

// XOR swizzles (ushort index into an LDS plane). The XOR only touches kd bits >= 2,
// so any 4-dword group whose base kd is a multiple of 4 stays 16B-contiguous/aligned.
__device__ inline int swz32(int row, int k) {   // row stride 32 ushorts (16 dwords)
    int kd = k >> 1; kd ^= (row & 3) << 2; return row * 32 + kd * 2 + (k & 1);
}
__device__ inline int swz64(int row, int k) {   // row stride 64 ushorts (32 dwords)
    int kd = k >> 1; kd ^= (row & 7) << 2; return row * 64 + kd * 2 + (k & 1);
}
__device__ inline int swz128(int row, int k) {  // row stride 128 ushorts (64 dwords)
    int kd = k >> 1; kd ^= (row & 7) << 3; return row * 128 + kd * 2 + (k & 1);
}

// Build one WMMA operand (8 dwords) from two 16B-aligned 16B chunks.
__device__ inline v8u ld_frag(const u16* p0, const u16* p1) {
    uint4 a = *(const uint4*)p0;
    uint4 b = *(const uint4*)p1;
    v8u r;
    r[0] = a.x; r[1] = a.y; r[2] = a.z; r[3] = a.w;
    r[4] = b.x; r[5] = b.y; r[6] = b.z; r[7] = b.w;
    return r;
}

__device__ inline v8f wmma_bf16(v8u a, v8u b, v8f c) {
    return __builtin_amdgcn_wmma_f32_16x16x32_bf16(
        false, __builtin_bit_cast(v16bf, a),
        false, __builtin_bit_cast(v16bf, b),
        (short)0, c, false, false);
}

// CDNA5 async global->LDS copy (ASYNCcnt); low 32 bits of a flat LDS pointer
// are the LDS address (ISA 10.2 aperture mapping), used directly as VDST.
__device__ inline void async_ld16(const void* gptr, void* lptr) {
    u32 l = (u32)(uintptr_t)lptr;
    u64 g = (u64)(uintptr_t)gptr;
    asm volatile("global_load_async_to_lds_b128 %0, %1, off"
                 :: "v"(l), "v"(g) : "memory");
}
__device__ inline void wait_async0() {
    asm volatile("s_wait_asynccnt 0" ::: "memory");
}

// ---- shared GEMM mainloop: C[128x128] tile of A[M,K](f32) @ B[K,N](f32) ---
// 256 threads = 8 waves, wave tile 64x32 (wm = w&1, wn = w>>1), BK = 32.
// Software-pipelined: global fetch of tile k+1 overlaps WMMA of tile k.
__device__ inline void gemm_tile(const float* __restrict__ A, const float* __restrict__ B,
                                 int lda, int ldb, int K, int bm, int bn,
                                 u16* Al, u16* Bl, v8f (&acc)[4][2]) {
    const int tid = threadIdx.x;
    const int w = tid >> 5, lane = tid & 31;
    const int wm = w & 1, wn = w >> 1;
    const int l15 = lane & 15, hi = lane >> 4, khalf = hi * 8;

    for (int i = 0; i < 4; ++i)
        for (int j = 0; j < 2; ++j)
            for (int e = 0; e < 8; ++e) acc[i][j][e] = 0.f;

    const int arow = tid >> 3,  acol = (tid & 7) * 4;    // A: 4 chunks, +32 rows apart
    const int brow = tid >> 5,  bcol = (tid & 31) * 4;   // B: 4 chunks, +8 k-rows apart

    float4 fa[4], fb[4];
    for (int i = 0; i < 4; ++i) {
        fa[i] = *(const float4*)(A + (size_t)(bm + arow + i * 32) * lda + acol);
        fb[i] = *(const float4*)(B + (size_t)(brow + i * 8) * ldb + bn + bcol);
    }

    for (int k0 = 0; k0 < K; k0 += 32) {
        // stage current tile regs -> LDS (bf16, swizzled)
        for (int i = 0; i < 4; ++i) {
            u32* d = (u32*)(Al + swz32(arow + i * 32, acol));
            d[0] = pack2(fa[i].x, fa[i].y);
            d[1] = pack2(fa[i].z, fa[i].w);
            Bl[swz32(bcol + 0, brow + i * 8)] = f2bf(fb[i].x);
            Bl[swz32(bcol + 1, brow + i * 8)] = f2bf(fb[i].y);
            Bl[swz32(bcol + 2, brow + i * 8)] = f2bf(fb[i].z);
            Bl[swz32(bcol + 3, brow + i * 8)] = f2bf(fb[i].w);
        }
        __syncthreads();

        const bool more = (k0 + 32 < K);
        if (more) {
            for (int i = 0; i < 4; ++i) {
                fa[i] = *(const float4*)(A + (size_t)(bm + arow + i * 32) * lda + (k0 + 32) + acol);
                fb[i] = *(const float4*)(B + (size_t)(k0 + 32 + brow + i * 8) * ldb + bn + bcol);
            }
        }

        v8u af[4], bfr[2];
        for (int ms = 0; ms < 4; ++ms) {
            int row = wm * 64 + ms * 16 + l15;
            af[ms] = ld_frag(Al + swz32(row, khalf), Al + swz32(row, 16 + khalf));
        }
        for (int ns = 0; ns < 2; ++ns) {
            int col = wn * 32 + ns * 16 + l15;
            bfr[ns] = ld_frag(Bl + swz32(col, khalf), Bl + swz32(col, 16 + khalf));
        }
        for (int ms = 0; ms < 4; ++ms)
            for (int ns = 0; ns < 2; ++ns)
                acc[ms][ns] = wmma_bf16(af[ms], bfr[ns], acc[ms][ns]);
        __syncthreads();
    }
}

// ---- kernel 1: qkv = x @ w_qkv ; scatter to q/k/v [B,H,N,64] bf16 (+ q f32)
// q is pre-scaled by hd^-0.5 in bf16 (scores use the unscaled f32 copy).
__global__ __launch_bounds__(256) void qkv_gemm_kernel(
    const float* __restrict__ x, const float* __restrict__ wqkv,
    u16* __restrict__ qbf, u16* __restrict__ kbf, u16* __restrict__ vbf,
    float* __restrict__ qf32) {
    __shared__ u16 Al[128 * 32];
    __shared__ u16 Bl[128 * 32];
    v8f acc[4][2];
    const int bn = blockIdx.x * 128, bm = blockIdx.y * 128;
    gemm_tile(x, wqkv, CC, 3 * CC, CC, bm, bn, Al, Bl, acc);

    const int tid = threadIdx.x, w = tid >> 5, lane = tid & 31;
    const int wm = w & 1, wn = w >> 1, l15 = lane & 15, hi = lane >> 4;
    for (int ms = 0; ms < 4; ++ms)
        for (int ns = 0; ns < 2; ++ns) {
            int ng = bn + wn * 32 + ns * 16 + l15;
            int t = ng / CC, jj = ng % CC;
            int h = jj >> 6, d = jj & 63;
            for (int r = 0; r < 8; ++r) {
                int mg = bm + wm * 64 + ms * 16 + r + hi * 8;
                int b = mg >> 11, n = mg & (NN - 1);
                float val = acc[ms][ns][r];
                size_t di = (((size_t)(b * NH + h)) * NN + n) * HD + d;
                if (t == 0) { qbf[di] = f2bf(val * 0.125f); qf32[di] = val; }
                else if (t == 1) kbf[di] = f2bf(val);
                else             vbf[di] = f2bf(val);
            }
        }
}

// ---- kernel 2: scores[b,h,n] = ||q||  (f32, exact w.r.t. our q)
__global__ __launch_bounds__(256) void scores_kernel(const float* __restrict__ qf32,
                                                     float* __restrict__ scores) {
    int t = blockIdx.x * 256 + threadIdx.x;     // < 98304 exactly
    const float4* p = (const float4*)(qf32 + (size_t)t * HD);
    float s = 0.f;
    for (int i = 0; i < 16; ++i) {
        float4 f = p[i];
        s += f.x * f.x + f.y * f.y + f.z * f.z + f.w * f.w;
    }
    scores[t] = sqrtf(s);
}

// ---- kernel 3: per (b,h) top-1024 of 2048 via bitonic sort (descending)
__global__ __launch_bounds__(1024) void topk_kernel(const float* __restrict__ scores,
                                                    int* __restrict__ keep_idx) {
    __shared__ float s[2 * KEEP];
    __shared__ int  si[2 * KEEP];
    const int bh = blockIdx.x, t = threadIdx.x;
    const float* sc = scores + (size_t)bh * NN;
    s[t] = sc[t];               si[t] = t;
    s[t + KEEP] = sc[t + KEEP]; si[t + KEEP] = t + KEEP;
    __syncthreads();
    for (int k = 2; k <= 2 * KEEP; k <<= 1)
        for (int j = k >> 1; j > 0; j >>= 1) {
            int i = 2 * t - (t & (j - 1));
            int ix = i + j;
            bool desc = ((i & k) == 0);
            float a = s[i], b = s[ix];
            bool sw = desc ? (a < b) : (a > b);
            if (sw) {
                s[i] = b; s[ix] = a;
                int tmp = si[i]; si[i] = si[ix]; si[ix] = tmp;
            }
            __syncthreads();
        }
    keep_idx[(size_t)bh * KEEP + t] = si[t];
}

// ---- kernel 4: flash attention over gathered keys/values -----------------
// block = 128 queries of one (b,h); 8 waves x 16 query rows; 8 key tiles of 128.
// LDS (48KB): KP region (32KB) = K tile (16KB) + V staging (16KB), later reused
// as the 8 P planes; Vt (16KB) = transposed V. K/V gathered with async-to-LDS.
__global__ __launch_bounds__(256) void attn_kernel(
    const u16* __restrict__ qbf, const u16* __restrict__ kbf,
    const u16* __restrict__ vbf, const int* __restrict__ keep_idx,
    float* __restrict__ attn_out) {
    __shared__ u16 KP[8 * 16 * 128];   // 32 KB
    __shared__ u16 Vt[64 * 128];       // 16 KB

    const int bh = blockIdx.y, qt = blockIdx.x;
    const int b = bh / NH, h = bh - b * NH;
    const int tid = threadIdx.x, w = tid >> 5, lane = tid & 31;
    const int l15 = lane & 15, hi = lane >> 4, khalf = hi * 8;

    // q fragments (A layout), two k-steps covering hd=64 (q pre-scaled by hd^-0.5)
    const u16* qrow = qbf + (((size_t)bh * NN) + qt * 128 + w * 16 + l15) * HD;
    v8u qf[2];
    for (int ks = 0; ks < 2; ++ks)
        qf[ks] = ld_frag(qrow + ks * 32 + khalf, qrow + ks * 32 + 16 + khalf);

    v8f Oacc[4];
    for (int i = 0; i < 4; ++i)
        for (int e = 0; e < 8; ++e) Oacc[i][e] = 0.f;
    float m_i[8], l_i[8];
    for (int r = 0; r < 8; ++r) { m_i[r] = -1e30f; l_i[r] = 0.f; }

    const int* kidx = keep_idx + (size_t)bh * KEEP;
    const u16* kb = kbf + (size_t)bh * NN * HD;
    const u16* vb = vbf + (size_t)bh * NN * HD;

    const int grow = tid >> 3, goff = (tid & 7) * 8;   // gather: 4 chunks, +32 rows apart

    for (int kt = 0; kt < 8; ++kt) {
        __syncthreads();   // previous tile's P / Vt / staging reads complete
        // async gather: K rows -> KP[0:8192], V rows -> staging KP[8192:16384]
        for (int i = 0; i < 4; ++i) {
            int row = grow + i * 32;
            int idx = kidx[kt * 128 + row];
            async_ld16(kb + (size_t)idx * HD + goff, KP + swz64(row, goff));
            async_ld16(vb + (size_t)idx * HD + goff, KP + 8192 + swz64(row, goff));
        }
        wait_async0();
        __syncthreads();   // all waves' async data landed in LDS

        // transpose V: staging [key][d] -> Vt [d][key]
        for (int i = 0; i < 4; ++i) {
            int row = grow + i * 32;
            uint4 t4 = *(const uint4*)(KP + 8192 + swz64(row, goff));
            const u16* vs = (const u16*)&t4;
            for (int j = 0; j < 8; ++j)
                Vt[swz128(goff + j, row)] = vs[j];
        }

        // S (16 x 128) = q @ K^T, via 8 n-subtiles x 2 k-steps
        v8f S[8];
        for (int ns = 0; ns < 8; ++ns) {
            for (int e = 0; e < 8; ++e) S[ns][e] = 0.f;
            int key = ns * 16 + l15;
            for (int ks = 0; ks < 2; ++ks) {
                v8u bfr = ld_frag(KP + swz64(key, ks * 32 + khalf),
                                  KP + swz64(key, ks * 32 + 16 + khalf));
                S[ns] = wmma_bf16(qf[ks], bfr, S[ns]);
            }
        }
        __syncthreads();   // K + staging dead, Vt writes visible; region reused for P

        // clip (scale already folded into q)
        for (int ns = 0; ns < 8; ++ns)
            for (int r = 0; r < 8; ++r)
                S[ns][r] = fminf(50.f, fmaxf(-50.f, S[ns][r]));
        // online softmax (rows live in 16-lane halves of the C layout)
        float alpha[8];
        for (int r = 0; r < 8; ++r) {
            float m = S[0][r];
            for (int ns = 1; ns < 8; ++ns) m = fmaxf(m, S[ns][r]);
            for (int o = 1; o < 16; o <<= 1) m = fmaxf(m, __shfl_xor(m, o, 32));
            float mn = fmaxf(m_i[r], m);
            alpha[r] = __expf(m_i[r] - mn);
            m_i[r] = mn;
        }
        for (int r = 0; r < 8; ++r) {
            float rs = 0.f;
            for (int ns = 0; ns < 8; ++ns) {
                float p = __expf(S[ns][r] - m_i[r]);
                S[ns][r] = p; rs += p;
            }
            for (int o = 1; o < 16; o <<= 1) rs += __shfl_xor(rs, o, 32);
            l_i[r] = l_i[r] * alpha[r] + rs;
            for (int ds = 0; ds < 4; ++ds) Oacc[ds][r] *= alpha[r];
        }
        // stage P (16x128) as bf16 into this wave's plane
        u16* Pw = KP + w * 2048;
        for (int ns = 0; ns < 8; ++ns)
            for (int r = 0; r < 8; ++r) {
                int m = r + hi * 8;
                Pw[swz128(m, ns * 16 + l15)] = f2bf(S[ns][r]);
            }
        // O += P @ V  (4 d-subtiles x 4 k-steps over 128 keys)
        for (int ds = 0; ds < 4; ++ds) {
            int dcol = ds * 16 + l15;
            for (int ks = 0; ks < 4; ++ks) {
                v8u af = ld_frag(Pw + swz128(l15, ks * 32 + khalf),
                                 Pw + swz128(l15, ks * 32 + 16 + khalf));
                v8u bv = ld_frag(Vt + swz128(dcol, ks * 32 + khalf),
                                 Vt + swz128(dcol, ks * 32 + 16 + khalf));
                Oacc[ds] = wmma_bf16(af, bv, Oacc[ds]);
            }
        }
    }
    // normalize and write f32 [B*N, 768]
    const int qg = qt * 128 + w * 16;
    for (int ds = 0; ds < 4; ++ds) {
        int col = h * HD + ds * 16 + l15;
        for (int r = 0; r < 8; ++r) {
            int m = r + hi * 8;
            float val = Oacc[ds][r] / l_i[r];
            attn_out[((size_t)(b * NN + qg + m)) * CC + col] = val;
        }
    }
}

// ---- kernel 5: out = clip(attn @ w_proj + b_proj, ±10) -------------------
__global__ __launch_bounds__(256) void proj_gemm_kernel(
    const float* __restrict__ attn, const float* __restrict__ wproj,
    const float* __restrict__ bproj, float* __restrict__ out) {
    __shared__ u16 Al[128 * 32];
    __shared__ u16 Bl[128 * 32];
    v8f acc[4][2];
    const int bn = blockIdx.x * 128, bm = blockIdx.y * 128;
    gemm_tile(attn, wproj, CC, CC, CC, bm, bn, Al, Bl, acc);

    const int tid = threadIdx.x, w = tid >> 5, lane = tid & 31;
    const int wm = w & 1, wn = w >> 1, l15 = lane & 15, hi = lane >> 4;
    for (int ms = 0; ms < 4; ++ms)
        for (int ns = 0; ns < 2; ++ns) {
            int ng = bn + wn * 32 + ns * 16 + l15;
            float bias = bproj[ng];
            for (int r = 0; r < 8; ++r) {
                int mg = bm + wm * 64 + ms * 16 + r + hi * 8;
                float val = acc[ms][ns][r] + bias;
                val = fminf(10.f, fmaxf(-10.f, val));
                out[(size_t)mg * CC + ng] = val;
            }
        }
}

// ---- host ----------------------------------------------------------------
extern "C" void kernel_launch(void* const* d_in, const int* in_sizes, int n_in,
                              void* d_out, int out_size, void* d_ws, size_t ws_size,
                              hipStream_t stream) {
    (void)in_sizes; (void)n_in; (void)out_size; (void)ws_size;
    const float* x      = (const float*)d_in[0];
    const float* w_qkv  = (const float*)d_in[1];
    const float* w_proj = (const float*)d_in[2];
    const float* b_proj = (const float*)d_in[3];
    float* out = (float*)d_out;

    char* ws = (char*)d_ws;
    size_t off = 0;
    auto carve = [&](size_t bytes) -> void* {
        void* p = ws + off;
        off += (bytes + 255) & ~(size_t)255;
        return p;
    };
    u16*   qbf    = (u16*)  carve(QKV_ELEMS * sizeof(u16));
    u16*   kbf    = (u16*)  carve(QKV_ELEMS * sizeof(u16));
    u16*   vbf    = (u16*)  carve(QKV_ELEMS * sizeof(u16));
    float* qf32   = (float*)carve(QKV_ELEMS * sizeof(float));
    float* scores = (float*)carve((size_t)BHN * sizeof(float));
    int*   kidx   = (int*)  carve((size_t)NB * NH * KEEP * sizeof(int));
    float* attn   = (float*)carve((size_t)NB * NN * CC * sizeof(float));

    qkv_gemm_kernel<<<dim3(3 * CC / 128, NB * NN / 128), 256, 0, stream>>>(
        x, w_qkv, qbf, kbf, vbf, qf32);
    scores_kernel<<<BHN / 256, 256, 0, stream>>>(qf32, scores);
    topk_kernel<<<NB * NH, 1024, 0, stream>>>(scores, kidx);
    attn_kernel<<<dim3(NN / 128, NB * NH), 256, 0, stream>>>(
        qbf, kbf, vbf, kidx, attn);
    proj_gemm_kernel<<<dim3(CC / 128, NB * NN / 128), 256, 0, stream>>>(
        attn, w_proj, b_proj, out);
}
